// MultiHead_Attention_31628139168433
// MI455X (gfx1250) — compile-verified, matrix operand order checked by reading, not run on hardware
//
#include <hip/hip_runtime.h>
#include <hip/hip_bf16.h>

// MI455X / gfx1250, wave32, WMMA f16 -> f32 accumulate.
//
// Pipeline (all on `stream`, scratch in d_ws, ~41 MB total, L2-resident):
//   K0 prep_weights : Wq/Wk/Wv -> f16 [H,F,D] (transposed), Wo -> f16 [D,G]
//   K1 proj x3      : Qh=q*Wq, Kh=k*Wk -> f16 [BH,S,F]; Vt=v*Wv -> f16 [BH,F,S]
//                     one wave = 16 x 64 output (A fragment reused 4x)
//   K2 stats        : per (bh,q) row: online max m_q and sum-exp l_q of scores
//   K3 atv          : out4[k,f] = sum_q exp(s[q,k]/8 - m_q)/l_q * V[q,f] (A^T V)
//                     one wave = 16 keys x all 64 f: scores + exp computed ONCE,
//                     P^T via LDS transpose, 4 accumulate WMMAs reuse it.
//                     Output stored directly in the reference's raw-reshape layout.
//   K4 outproj      : Out = Ofused @ Wo (f32), one wave = 16 x 64 output

typedef __attribute__((ext_vector_type(16))) _Float16 v16h;
typedef __attribute__((ext_vector_type(8)))  _Float16 v8h;
typedef __attribute__((ext_vector_type(8)))  float    v8f;

#define B_ 2
#define S_ 2048
#define D_ 1024
#define H_ 16
#define F_ 64
#define G_ (H_ * F_)   // 1024
#define WAVES_ 4       // waves per block (blockDim = 32 x 4)

static __device__ __forceinline__ v16h pack16(v8h lo, v8h hi) {
  v16h a;
#pragma unroll
  for (int i = 0; i < 8; i++) { a[i] = lo[i]; a[i + 8] = hi[i]; }
  return a;
}

// A/B fragment load, f16 row-major source with leading dim `ld` (halves).
// ISA 16-bit 16x32 layout: lanes 0-15 hold line=lane, K={0..7,16..23};
// lanes 16-31 hold line=lane-16, K={8..15,24..31}. Two contiguous 16B loads.
static __device__ __forceinline__ v16h load_frag_f16(const _Float16* base, int ld) {
  const int lane = threadIdx.x;
  const int line = lane & 15;
  const int k0   = (lane >> 4) << 3;  // 0 or 8
  const _Float16* p = base + (size_t)line * ld + k0;
  v8h lo = *(const v8h*)(p);
  v8h hi = *(const v8h*)(p + 16);
  return pack16(lo, hi);
}

// Same, but from f32 source with on-the-fly f16 convert (two 32B loads).
static __device__ __forceinline__ v16h load_frag_f32(const float* base, int ld) {
  const int lane = threadIdx.x;
  const int line = lane & 15;
  const int k0   = (lane >> 4) << 3;
  const float* p = base + (size_t)line * ld + k0;
  v8f lo = *(const v8f*)(p);
  v8f hi = *(const v8f*)(p + 16);
  v16h a;
#pragma unroll
  for (int i = 0; i < 8; i++) { a[i] = (_Float16)lo[i]; a[i + 8] = (_Float16)hi[i]; }
  return a;
}

static __device__ __forceinline__ v8f wmma16(v16h a, v16h b, v8f c) {
  return __builtin_amdgcn_wmma_f32_16x16x32_f16(false, a, false, b, (short)0, c,
                                                false, false);
}

// ---------------- K0: weight prep (f32 -> f16, transpose) ----------------
// t in [0, H*D*F) == [0, D*G): WxT[h][f][d] = Wx[h][d][f]; WoT[d][g] = Wo[g][d].
__global__ void mha_prep_weights(const float* __restrict__ Wq, const float* __restrict__ Wk,
                                 const float* __restrict__ Wv, const float* __restrict__ Wo,
                                 _Float16* __restrict__ WqT, _Float16* __restrict__ WkT,
                                 _Float16* __restrict__ WvT, _Float16* __restrict__ WoT) {
  const int t = blockIdx.x * blockDim.x + threadIdx.x;  // 0 .. 1048575
  const int d = t % D_;
  const int f = (t / D_) % F_;
  const int h = t / (D_ * F_);
  const size_t src = (size_t)(h * D_ + d) * F_ + f;
  WqT[t] = (_Float16)Wq[src];
  WkT[t] = (_Float16)Wk[src];
  WvT[t] = (_Float16)Wv[src];
  const int g  = t % G_;
  const int d2 = t / G_;
  WoT[t] = (_Float16)Wo[(size_t)g * D_ + d2];
}

// ---------------- K1: head projection  Y[bh] = X[b] (SxD) * W[h] (DxF) -----
// One wave per (bh, 16 s-rows): computes full 16 x F_ strip, A reused 4x.
// transpose_out=0: Y f16 [BH,S,F];  transpose_out=1: Y f16 [BH,F,S] (for V).
__global__ void mha_proj(const float* __restrict__ X, const _Float16* __restrict__ WT,
                         _Float16* __restrict__ Y, int transpose_out) {
  const int w  = blockIdx.x * WAVES_ + threadIdx.y;  // 4096 waves
  const int sT = w & 127;
  const int bh = w >> 7;
  const int b  = bh >> 4;

  const float*    Abase = X  + ((size_t)(b * S_) + sT * 16) * D_;
  const _Float16* Bbase = WT + (size_t)((bh & 15) * F_) * D_;

  v8f c[4] = {};
  for (int kk = 0; kk < D_; kk += 32) {
    v16h a = load_frag_f32(Abase + kk, D_);
#pragma unroll
    for (int j = 0; j < 4; j++) {
      v16h bm = load_frag_f16(Bbase + (size_t)(j * 16) * D_ + kk, D_);
      c[j] = wmma16(a, bm, c[j]);
    }
  }

  const int lane = threadIdx.x;
  const int n    = lane & 15;
  const int mb   = (lane >> 4) << 3;  // C/D layout: VGPR r -> row r (+8 for hi half)
  if (!transpose_out) {
#pragma unroll
    for (int j = 0; j < 4; j++) {
      _Float16* out = Y + ((size_t)bh * S_ + sT * 16) * F_ + j * 16 + n;
#pragma unroll
      for (int r = 0; r < 8; r++) out[(size_t)(mb + r) * F_] = (_Float16)c[j][r];
    }
  } else {
    // Vt[bh][f][s]: per lane f fixed, s = sT*16 + mb + r -> contiguous 16B store
#pragma unroll
    for (int j = 0; j < 4; j++) {
      _Float16* out = Y + ((size_t)bh * F_ + j * 16 + n) * S_ + sT * 16 + mb;
      v8h pk;
#pragma unroll
      for (int r = 0; r < 8; r++) pk[r] = (_Float16)c[j][r];
      *(v8h*)out = pk;
    }
  }
}

// ---------------- K2: per-row softmax stats over scores = QK^T/8 -----------
// One wave per 16 query rows; online (m, l) over all S keys, then lane-reduce.
__global__ void mha_stats(const _Float16* __restrict__ Qh, const _Float16* __restrict__ Kh,
                          float* __restrict__ Mrow, float* __restrict__ Linv) {
  const int w  = blockIdx.x * WAVES_ + threadIdx.y;  // 4096 waves
  const int qT = w & 127;
  const int bh = w >> 7;

  const _Float16* Abase = Qh + ((size_t)bh * S_ + qT * 16) * F_;
  const v16h a0 = load_frag_f16(Abase, F_);
  const v16h a1 = load_frag_f16(Abase + 32, F_);

  float m[8], l[8];
#pragma unroll
  for (int i = 0; i < 8; i++) { m[i] = -1e30f; l[i] = 0.0f; }
  const float scale = 0.125f;  // 1/sqrt(F)

  for (int kt = 0; kt < S_ / 16; kt++) {
    const _Float16* Bbase = Kh + ((size_t)bh * S_ + kt * 16) * F_;
    v16h b0 = load_frag_f16(Bbase, F_);
    v16h b1 = load_frag_f16(Bbase + 32, F_);
    v8f c = {};
    c = wmma16(a0, b0, c);
    c = wmma16(a1, b1, c);
#pragma unroll
    for (int i = 0; i < 8; i++) {
      float v  = c[i] * scale;
      float nm = fmaxf(m[i], v);
      l[i] = l[i] * __expf(m[i] - nm) + __expf(v - nm);
      m[i] = nm;
    }
  }
  // Reduce across the 16 lanes of each half-wave (xor 1,2,4,8 stays in-half).
#pragma unroll
  for (int off = 1; off < 16; off <<= 1) {
#pragma unroll
    for (int i = 0; i < 8; i++) {
      float om = __shfl_xor(m[i], off, 32);
      float ol = __shfl_xor(l[i], off, 32);
      float nm = fmaxf(m[i], om);
      l[i] = l[i] * __expf(m[i] - nm) + ol * __expf(om - nm);
      m[i] = nm;
    }
  }
  const int lane = threadIdx.x;
  if ((lane & 15) == 0) {
    const int rb = (lane >> 4) << 3;
    const size_t base = (size_t)bh * S_ + qT * 16 + rb;
#pragma unroll
    for (int i = 0; i < 8; i++) { Mrow[base + i] = m[i]; Linv[base + i] = 1.0f / l[i]; }
  }
}

// ---------------- K3: out4 = attn^T V, streamed over q, stored reshaped ----
// One wave per (bh, 16-key tile): scores/exp computed once, P^T reused for
// all 4 f-tiles (16 x 64 output strip per wave).
__global__ void mha_atv(const _Float16* __restrict__ Qh, const _Float16* __restrict__ Kh,
                        const _Float16* __restrict__ Vt, const float* __restrict__ Mrow,
                        const float* __restrict__ Linv, _Float16* __restrict__ Ofused) {
  __shared__ _Float16 plds_all[WAVES_][16 * 40];  // [k-line][q], row stride 40 halves
  _Float16* plds = plds_all[threadIdx.y];

  const int w  = blockIdx.x * WAVES_ + threadIdx.y;  // 4096 waves
  const int kT = w & 127;
  const int bh = w >> 7;
  const int lane = threadIdx.x;
  const int line = lane & 15;
  const int half = lane >> 4;

  // B-fragment for score WMMA: K^T columns of this key tile (fixed for loop).
  const _Float16* Kbase = Kh + ((size_t)bh * S_ + kT * 16) * F_;
  const v16h kb0 = load_frag_f16(Kbase, F_);
  const v16h kb1 = load_frag_f16(Kbase + 32, F_);

  const _Float16* Vbase = Vt + (size_t)bh * F_ * S_;
  const float* mptr = Mrow + (size_t)bh * S_;
  const float* lptr = Linv + (size_t)bh * S_;

  v8f acc[4] = {};
  const float scale = 0.125f;

  for (int qb = 0; qb < S_; qb += 32) {
    // scores for q-block [qb, qb+32) x key tile (two 16x16 WMMA tiles)
    const _Float16* Qbase = Qh + ((size_t)bh * S_ + qb) * F_;
    __builtin_prefetch(Qbase + (size_t)32 * F_, 0, 1);  // next q-block
    v16h a00 = load_frag_f16(Qbase, F_);
    v16h a01 = load_frag_f16(Qbase + 32, F_);
    v16h a10 = load_frag_f16(Qbase + (size_t)16 * F_, F_);
    v16h a11 = load_frag_f16(Qbase + (size_t)16 * F_ + 32, F_);
    v8f s0 = {}; s0 = wmma16(a00, kb0, s0); s0 = wmma16(a01, kb1, s0);
    v8f s1 = {}; s1 = wmma16(a10, kb0, s1); s1 = wmma16(a11, kb1, s1);

    // per-q-row softmax stats (rows mb..mb+7 of each sub-tile)
    v8f m0 = *(const v8f*)(mptr + qb + half * 8);
    v8f l0 = *(const v8f*)(lptr + qb + half * 8);
    v8f m1 = *(const v8f*)(mptr + qb + 16 + half * 8);
    v8f l1 = *(const v8f*)(lptr + qb + 16 + half * 8);

    v8h p0, p1;
#pragma unroll
    for (int i = 0; i < 8; i++) {
      p0[i] = (_Float16)(__expf(s0[i] * scale - m0[i]) * l0[i]);
      p1[i] = (_Float16)(__expf(s1[i] * scale - m1[i]) * l1[i]);
    }
    // transpose P (q x k) -> P^T (k x q) through LDS; all 16B accesses
    *(v8h*)(plds + line * 40 + half * 8)      = p0;  // q = half*8 + i
    *(v8h*)(plds + line * 40 + 16 + half * 8) = p1;  // q = 16 + half*8 + i
    __builtin_amdgcn_wave_barrier();                 // order vs loads (LDS in-order)
    const int k0 = half * 8;
    v8h alo = *(const v8h*)(plds + line * 40 + k0);
    v8h ahi = *(const v8h*)(plds + line * 40 + k0 + 16);
    v16h aP = pack16(alo, ahi);
    __builtin_amdgcn_wave_barrier();                 // order next-iter stores vs loads

    // B-fragments: V rows qb..qb+31 for each f tile (Vt is [f][s], contiguous)
#pragma unroll
    for (int j = 0; j < 4; j++) {
      v16h bV = load_frag_f16(Vbase + (size_t)(j * 16) * S_ + qb, S_);
      acc[j] = wmma16(aP, bV, acc[j]);
    }
  }

  // Raw-reshape store: Ofused[b][h*128 + kT][(k%16)*64 + f]
  const int b = bh >> 4, h = bh & 15;
  const int mb = half * 8;
#pragma unroll
  for (int j = 0; j < 4; j++) {
    _Float16* out = Ofused + ((size_t)(b * S_ + h * 128 + kT)) * G_ + (j * 16 + line);
#pragma unroll
    for (int r = 0; r < 8; r++) out[(size_t)(mb + r) * 64] = (_Float16)acc[j][r];
  }
}

// ---------------- K4: Out = Ofused (S x G) @ Wo (G x D), f32 out ----------
// One wave per (b, 16 s-rows, group of 4 d-tiles): 16 x 64 strip, A reused 4x.
__global__ void mha_outproj(const _Float16* __restrict__ Ofused,
                            const _Float16* __restrict__ WoT, float* __restrict__ Out) {
  const int w  = blockIdx.x * WAVES_ + threadIdx.y;  // 4096 waves
  const int dG = w & 15;           // group of 4 d-tiles
  const int sT = (w >> 4) & 127;
  const int b  = w >> 11;

  const _Float16* Abase = Ofused + ((size_t)(b * S_) + sT * 16) * G_;
  const _Float16* Bbase = WoT + (size_t)(dG * 64) * G_;
  v8f c[4] = {};
  for (int g = 0; g < G_; g += 32) {
    v16h a = load_frag_f16(Abase + g, G_);
#pragma unroll
    for (int j = 0; j < 4; j++) {
      v16h bm = load_frag_f16(Bbase + (size_t)(j * 16) * G_ + g, G_);
      c[j] = wmma16(a, bm, c[j]);
    }
  }
  const int lane = threadIdx.x;
  const int n    = lane & 15;
  const int mb   = (lane >> 4) << 3;
#pragma unroll
  for (int j = 0; j < 4; j++) {
    float* out = Out + ((size_t)(b * S_) + sT * 16) * D_ + dG * 64 + j * 16 + n;
#pragma unroll
    for (int r = 0; r < 8; r++) out[(size_t)(mb + r) * D_] = c[j][r];
  }
}

extern "C" void kernel_launch(void* const* d_in, const int* in_sizes, int n_in,
                              void* d_out, int out_size, void* d_ws, size_t ws_size,
                              hipStream_t stream) {
  (void)in_sizes; (void)n_in; (void)out_size; (void)ws_size;
  const float* q  = (const float*)d_in[0];
  const float* k  = (const float*)d_in[1];
  const float* v  = (const float*)d_in[2];
  const float* Wq = (const float*)d_in[3];
  const float* Wk = (const float*)d_in[4];
  const float* Wv = (const float*)d_in[5];
  const float* Wo = (const float*)d_in[6];
  float* out = (float*)d_out;

  size_t off = 0;
  auto alloc = [&](size_t bytes) -> char* {
    char* p = (char*)d_ws + off;
    off += (bytes + 255) & ~(size_t)255;
    return p;
  };
  _Float16* WqT = (_Float16*)alloc((size_t)H_ * D_ * F_ * 2);   // 2 MB
  _Float16* WkT = (_Float16*)alloc((size_t)H_ * D_ * F_ * 2);
  _Float16* WvT = (_Float16*)alloc((size_t)H_ * D_ * F_ * 2);
  _Float16* WoT = (_Float16*)alloc((size_t)G_ * D_ * 2);
  _Float16* Qh  = (_Float16*)alloc((size_t)B_ * H_ * S_ * F_ * 2);  // 8 MB
  _Float16* Kh  = (_Float16*)alloc((size_t)B_ * H_ * S_ * F_ * 2);
  _Float16* Vt  = (_Float16*)alloc((size_t)B_ * H_ * F_ * S_ * 2);
  _Float16* Of  = (_Float16*)alloc((size_t)B_ * S_ * G_ * 2);
  float*    Mr  = (float*)alloc((size_t)B_ * H_ * S_ * 4);          // 256 KB
  float*    Li  = (float*)alloc((size_t)B_ * H_ * S_ * 4);
  // total ~41 MB of workspace

  dim3 blk(32, WAVES_);
  mha_prep_weights<<<(H_ * D_ * F_) / 256, 256, 0, stream>>>(Wq, Wk, Wv, Wo,
                                                             WqT, WkT, WvT, WoT);
  mha_proj<<<4096 / WAVES_, blk, 0, stream>>>(q, WqT, Qh, 0);
  mha_proj<<<4096 / WAVES_, blk, 0, stream>>>(k, WkT, Kh, 0);
  mha_proj<<<4096 / WAVES_, blk, 0, stream>>>(v, WvT, Vt, 1);
  mha_stats<<<4096 / WAVES_, blk, 0, stream>>>(Qh, Kh, Mr, Li);
  mha_atv<<<4096 / WAVES_, blk, 0, stream>>>(Qh, Kh, Vt, Mr, Li, Of);
  mha_outproj<<<4096 / WAVES_, blk, 0, stream>>>(Of, WoT, out);
}
